// TRLinear_48962627174848
// MI455X (gfx1250) — matrix operand order repaired and checked by reading, not compiled, not run on hardware
//
#include <hip/hip_runtime.h>

// ---------------------------------------------------------------------------
// TR-linear:  out = x @ W + bias,   W reconstructed from tensor-ring cores.
// Strategy: reconstruct W^T in bf16 into workspace (tiny FLOPs, 8 MB, stays in
// L2), convert x to bf16 once, then a WMMA bf16 GEMM (68.7 GFLOP hot part)
// with fp32 accumulation, double-buffered LDS fed by CDNA5 async
// global->LDS loads (ASYNCcnt path, no VGPR staging).
// ---------------------------------------------------------------------------

typedef __attribute__((ext_vector_type(16))) __bf16 bf16x16;
typedef __attribute__((ext_vector_type(8)))  __bf16 bf16x8;
typedef __attribute__((ext_vector_type(8)))  float  f32x8;

#define BATCH 8192
#define IN_F  1024
#define OUT_F 4096
#define RANK  8

// round-to-nearest-even fp32 -> bf16
__device__ __forceinline__ unsigned short f32_to_bf16(float f) {
    unsigned int u = __float_as_uint(f);
    unsigned int r = u + 0x7FFFu + ((u >> 16) & 1u);
    return (unsigned short)(r >> 16);
}

// CDNA5 async global->LDS copy, GVS mode: saddr = 64-bit SGPR base,
// vaddr = per-lane 32-bit byte offset, vdst = LDS byte address.
// Per ISA 10.2, the LDS byte address is the low 32 bits of the flat address.
__device__ __forceinline__ void async_ld_b128(void* lds_ptr, unsigned voff,
                                              const void* sbase) {
    unsigned l = (unsigned)(uintptr_t)lds_ptr;
    asm volatile("global_load_async_to_lds_b128 %0, %1, %2"
                 :: "v"(l), "v"(voff), "s"(sbase) : "memory");
}
#define ASYNC_WAIT_4() asm volatile("s_wait_asynccnt 0x4" ::: "memory")
#define ASYNC_WAIT_0() asm volatile("s_wait_asynccnt 0x0" ::: "memory")

// ---------------------------------------------------------------------------
// Stage 1: g23[b,r,s,t,u,a] = sum_c g2[b,r,s,c] * g3[c,t,u,a]
// layout: ((((b*16+r)*16+s)*8+t)*16+u)*8+a  -> 2,097,152 fp32 (8 MiB)
// ---------------------------------------------------------------------------
__global__ __launch_bounds__(256) void build_g23_kernel(
    const float* __restrict__ g2, const float* __restrict__ g3,
    float* __restrict__ g23)
{
    int idx = blockIdx.x * 256 + threadIdx.x;   // 0 .. 2^21-1
    int a = idx & 7;
    int u = (idx >> 3) & 15;
    int t = (idx >> 7) & 7;
    int s = (idx >> 10) & 15;
    int r = (idx >> 14) & 15;
    int b = idx >> 18;
    float acc = 0.f;
#pragma unroll
    for (int c = 0; c < RANK; ++c) {
        float v2 = g2[((b * 16 + r) * 16 + s) * 8 + c];
        float v3 = g3[((c * 8 + t) * 16 + u) * 8 + a];
        acc = fmaf(v2, v3, acc);
    }
    g23[idx] = acc;
}

// ---------------------------------------------------------------------------
// Stage 2: Wt[n][k] (bf16, N-major, K contiguous)
//   n = q*256 + s*16 + u ; k = p*128 + r*8 + t
//   Wt[n,k] = sum_{a,b} g1[a,p,q,b] * g23[b,r,s,t,u,a]
// ---------------------------------------------------------------------------
__global__ __launch_bounds__(256) void build_Wt_kernel(
    const float* __restrict__ g1, const float* __restrict__ g23,
    unsigned short* __restrict__ Wt)
{
    int idx = blockIdx.x * 256 + threadIdx.x;   // 0 .. 4M-1
    int k = idx & 1023;
    int n = idx >> 10;
    int p = k >> 7, r = (k >> 3) & 15, t = k & 7;
    int q = n >> 8, s = (n >> 4) & 15, u = n & 15;
    float acc = 0.f;
#pragma unroll
    for (int b = 0; b < RANK; ++b) {
        const float* g23p = &g23[((((b * 16 + r) * 16 + s) * 8 + t) * 16 + u) * 8];
#pragma unroll
        for (int a = 0; a < RANK; ++a) {
            float v1 = g1[((a * 8 + p) * 16 + q) * 8 + b];
            acc = fmaf(v1, g23p[a], acc);
        }
    }
    Wt[idx] = f32_to_bf16(acc);
}

// ---------------------------------------------------------------------------
// Stage 3: x fp32 -> bf16 (vectorized x4)
// ---------------------------------------------------------------------------
__global__ __launch_bounds__(256) void convert_x_kernel(
    const float* __restrict__ x, unsigned short* __restrict__ xb)
{
    int idx = blockIdx.x * 256 + threadIdx.x;       // one float4 each
    const float4 v = *(const float4*)(x + (size_t)idx * 4);
    ushort4 o;
    o.x = f32_to_bf16(v.x);
    o.y = f32_to_bf16(v.y);
    o.z = f32_to_bf16(v.z);
    o.w = f32_to_bf16(v.w);
    *(ushort4*)(xb + (size_t)idx * 4) = o;
}

// ---------------------------------------------------------------------------
// Stage 4: GEMM  out[8192,4096] = xb[8192,1024] * Wt^T + bias
//   block tile 128x128, BK=32, 256 threads = 8 waves (wave32)
//   wave tile 32x64 = 2x4 fragments of v_wmma_f32_16x16x32_bf16
//   double-buffered LDS filled by global_load_async_to_lds_b128 (ASYNCcnt),
//   manually unrolled x2 so both pipeline halves have identical register flow
// ---------------------------------------------------------------------------
#define BM 128
#define BN 128
#define BK 32
#define KTILES (IN_F / BK)   // 32

__global__ __launch_bounds__(256) void tr_gemm_wmma_kernel(
    const unsigned short* __restrict__ xb,   // [8192][1024] bf16
    const unsigned short* __restrict__ Wt,   // [4096][1024] bf16 (N-major)
    const float* __restrict__ bias,          // [4096]
    float* __restrict__ out)                 // [8192][4096] fp32
{
    // double-buffered tiles: A [2][BM][BK], B [2][BN][BK] (bf16) = 32 KiB
    __shared__ uint4 As4[2][(BM * BK) / 8];
    __shared__ uint4 Bs4[2][(BN * BK) / 8];

    const int tid  = threadIdx.x;
    const int lane = tid & 31;
    const int warp = tid >> 5;               // 0..7
    const int wm0  = (warp >> 1) * 32;       // 0,32,64,96
    const int wn0  = (warp & 1) * 64;        // 0,64

    const int m0 = blockIdx.y * BM;
    const int n0 = blockIdx.x * BN;

    const int lane16 = lane & 15;            // row/col within fragment
    const int half   = lane >> 4;            // 0: lanes 0-15, 1: lanes 16-31

    // per-thread static byte offsets for the 4 async b128 tile loads
    // uint4 index idx -> element e=idx*8, row=idx>>2, col=(idx&3)*8
    const int rowL = tid >> 2;               // 0..63 (second load: +64)
    const int colL = (tid & 3) * 8;
    const unsigned voffA0 = (unsigned)(((m0 + rowL) * IN_F + colL) * 2);
    const unsigned voffA1 = voffA0 + 64u * IN_F * 2u;
    const unsigned voffB0 = (unsigned)(((n0 + rowL) * IN_F + colL) * 2);
    const unsigned voffB1 = voffB0 + 64u * IN_F * 2u;

    f32x8 acc[2][4];
#pragma unroll
    for (int fm = 0; fm < 2; ++fm)
#pragma unroll
        for (int fn = 0; fn < 4; ++fn)
            acc[fm][fn] = (f32x8)(0.0f);

    auto load_tiles = [&](int buf, int ktn) {
        const unsigned kb = (unsigned)(ktn * 2);
        async_ld_b128(&As4[buf][tid],       voffA0 + kb, xb);
        async_ld_b128(&As4[buf][tid + 256], voffA1 + kb, xb);
        async_ld_b128(&Bs4[buf][tid],       voffB0 + kb, Wt);
        async_ld_b128(&Bs4[buf][tid + 256], voffB1 + kb, Wt);
    };

    auto compute_tile = [&](int buf) {
        const __bf16* As = (const __bf16*)&As4[buf][0];
        const __bf16* Bs = (const __bf16*)&Bs4[buf][0];

        // A fragments: this code's contiguous k order (k = 16*half + s), same as B
        //   lanes 0-15 : M=lane,    halves 0-15 = K 0..15
        //   lanes 16-31: M=lane-16, halves 0-15 = K 16..31
        bf16x16 afrag[2];
#pragma unroll
        for (int fm = 0; fm < 2; ++fm) {
            int row = wm0 + fm * 16 + lane16;
            int kb  = half * 16;
            afrag[fm] = *(const bf16x16*)(As + row * BK + kb);
        }

        // B fragments: this code's contiguous k order (k = 16*half + s), same as A
        //   lanes 0-15 : N=lane,    halves 0-15 = K 0..15
        //   lanes 16-31: N=lane-16, halves 0-15 = K 16..31
        bf16x16 bfrag[4];
#pragma unroll
        for (int fn = 0; fn < 4; ++fn) {
            int col = wn0 + fn * 16 + lane16;
            int kb  = half * 16;
            bfrag[fn] = *(const bf16x16*)(Bs + col * BK + kb);
        }

#pragma unroll
        for (int fm = 0; fm < 2; ++fm)
#pragma unroll
            for (int fn = 0; fn < 4; ++fn)
                acc[fm][fn] = __builtin_amdgcn_wmma_f32_16x16x32_bf16(
                    false, afrag[fm], false, bfrag[fn],
                    (short)0, acc[fm][fn], false, false);
    };

    // -------- software pipeline, manually unrolled x2 (symmetric halves) ----
    // 32 K-tiles: loop handles tiles 0..29 (15 double-steps), tail does 30,31.
    load_tiles(0, 0);
#pragma unroll 1
    for (int kt = 0; kt < (KTILES - 2) * BK; kt += 2 * BK) {
        // half A: compute buf0 (tile kt/BK), prefetch buf1 (tile +1)
        load_tiles(1, kt + BK);
        ASYNC_WAIT_4();          // retire the oldest tile's 4 loads
        __syncthreads();
        compute_tile(0);
        __syncthreads();
        // half B: compute buf1 (tile +1), prefetch buf0 (tile +2)
        load_tiles(0, kt + 2 * BK);
        ASYNC_WAIT_4();
        __syncthreads();
        compute_tile(1);
        __syncthreads();
    }
    // tail: tile 30 is in flight to buf0; issue tile 31 -> buf1
    load_tiles(1, (KTILES - 1) * BK);
    ASYNC_WAIT_4();
    __syncthreads();
    compute_tile(0);             // tile 30
    __syncthreads();
    ASYNC_WAIT_0();
    __syncthreads();
    compute_tile(1);             // tile 31

    // C/D layout: VGPR i -> M = i (lanes 0-15) or i+8 (lanes 16-31), N = lane&15
#pragma unroll
    for (int fm = 0; fm < 2; ++fm) {
        int rbase = m0 + wm0 + fm * 16 + half * 8;
#pragma unroll
        for (int fn = 0; fn < 4; ++fn) {
            int col = n0 + wn0 + fn * 16 + lane16;
            float bv = bias[col];
#pragma unroll
            for (int i = 0; i < 8; ++i)
                out[(size_t)(rbase + i) * OUT_F + col] = acc[fm][fn][i] + bv;
        }
    }
}

// ---------------------------------------------------------------------------
extern "C" void kernel_launch(void* const* d_in, const int* in_sizes, int n_in,
                              void* d_out, int out_size, void* d_ws, size_t ws_size,
                              hipStream_t stream) {
    const float* x    = (const float*)d_in[0];
    const float* g1   = (const float*)d_in[1];
    const float* g2   = (const float*)d_in[2];
    const float* g3   = (const float*)d_in[3];
    const float* bias = (const float*)d_in[4];
    float* out = (float*)d_out;

    char* ws = (char*)d_ws;
    float*          g23 = (float*)ws;                              // 8 MiB fp32
    unsigned short* Wt  = (unsigned short*)(ws + (8u << 20));      // 8 MiB bf16
    unsigned short* xb  = (unsigned short*)(ws + (16u << 20));     // 16 MiB bf16

    // Stage 1: g23 = g2 (x) g3   (2,097,152 elems)
    build_g23_kernel<<<dim3(2097152 / 256), dim3(256), 0, stream>>>(g2, g3, g23);
    // Stage 2: Wt (bf16, N-major) (4,194,304 elems)
    build_Wt_kernel<<<dim3(4194304 / 256), dim3(256), 0, stream>>>(g1, g23, Wt);
    // Stage 3: x -> bf16 (8,388,608 elems, 4 per thread)
    convert_x_kernel<<<dim3(2097152 / 256), dim3(256), 0, stream>>>(x, xb);
    // Stage 4: WMMA GEMM, grid = (4096/128, 8192/128) = (32, 64)
    tr_gemm_wmma_kernel<<<dim3(OUT_F / BN, BATCH / BM), dim3(256), 0, stream>>>(
        xb, Wt, bias, out);
}
